// CrossAttention_49417893707843
// MI455X (gfx1250) — matrix operand order repaired and checked by reading, not compiled, not run on hardware
//
#include <hip/hip_runtime.h>

typedef __bf16 bf16;
typedef __attribute__((ext_vector_type(16))) __bf16 v16bf;
typedef __attribute__((ext_vector_type(4)))  __bf16 bf16x4;
typedef __attribute__((ext_vector_type(8)))  float  v8f;
typedef __attribute__((ext_vector_type(4)))  float  f32x4;
typedef __attribute__((ext_vector_type(4)))  unsigned int u32x4;

// exact pointee type expected by the async-to-LDS builtin (v4i32)
typedef int v4i __attribute__((vector_size(16)));
typedef __attribute__((address_space(1))) v4i* gptr_v4i;
typedef __attribute__((address_space(3))) v4i* lptr_v4i;

union Frag16 { v16bf v; u32x4 q[2]; };

static constexpr int BATCH = 4;
static constexpr int LQ    = 2048;
static constexpr int LK    = 2048;
static constexpr int DIM   = 1024;
static constexpr int H     = 16;
static constexpr int D     = 64;
static constexpr int AD    = H * D;       // 1024

#if __has_builtin(__builtin_amdgcn_global_load_async_to_lds_b128)
#define USE_ASYNC_LDS 1
#else
#define USE_ASYNC_LDS 0
#endif

// 16-byte global -> LDS copy; async (no VGPR staging, ASYNCcnt) when available.
__device__ inline void copy16_g2l(bf16* lds_dst, const bf16* gsrc) {
#if USE_ASYNC_LDS
  __builtin_amdgcn_global_load_async_to_lds_b128(
      (gptr_v4i)(void*)gsrc, (lptr_v4i)(void*)lds_dst, 0, 0);
#else
  *(u32x4*)lds_dst = *(const u32x4*)gsrc;
#endif
}

__device__ inline void wait_async_lds() {
#if USE_ASYNC_LDS
#if __has_builtin(__builtin_amdgcn_s_wait_asynccnt)
  __builtin_amdgcn_s_wait_asynccnt(0);
#else
  asm volatile("s_wait_asynccnt 0x0" ::: "memory");
#endif
#endif
}

__device__ inline v8f v8f_zero() {
  v8f z;
#pragma unroll
  for (int i = 0; i < 8; i++) z[i] = 0.0f;
  return z;
}

__device__ inline v8f wmma_bf16(const Frag16& a, const Frag16& b, v8f c) {
  // D = A(16x32 bf16) * B(32x16 bf16) + C(16x16 f32)
  return __builtin_amdgcn_wmma_f32_16x16x32_bf16(
      /*neg_a=*/false, a.v, /*neg_b=*/false, b.v,
      /*c_mod=*/(short)0, c, /*reuse_a=*/false, /*reuse_b=*/false);
}

// ---------------------------------------------------------------------------
// Stage 0: f32 -> bf16 conversion (vectorized 4-wide)
// ---------------------------------------------------------------------------
__global__ void cvt_f32_to_bf16(const f32x4* __restrict__ in,
                                bf16x4* __restrict__ out, int n4) {
  int i = blockIdx.x * blockDim.x + threadIdx.x;
  int stride = gridDim.x * blockDim.x;
  for (; i < n4; i += stride) {
    f32x4 a = in[i];
    out[i] = __builtin_convertvector(a, bf16x4);
  }
}

// ---------------------------------------------------------------------------
// Stage 1/3: tiled bf16 WMMA GEMM:  Out[M,N] = A[M,K] @ B[K,N] + bias[N]
// Block tile 128(M) x 64(N), K-step 32, 256 threads = 8 waves.
// Double-buffered LDS; A tile via async-to-LDS, B tile transposed to N-major.
// ---------------------------------------------------------------------------
template <bool OUT_F32>
__global__ __launch_bounds__(256) void gemm_bf16(
    const bf16* __restrict__ A, const bf16* __restrict__ B,
    const float* __restrict__ bias, void* __restrict__ outv,
    int M, int N, int K) {
  __shared__ bf16 sA[2][128 * 32];
  __shared__ bf16 sB[2][64 * 32];

  const int t = threadIdx.x;
  const int w = t >> 5;
  const int lane = t & 31;
  const int half = lane >> 4;
  const int l16 = lane & 15;
  const int m0 = blockIdx.y * 128;
  const int n0 = blockIdx.x * 64;

  v8f acc[4];
#pragma unroll
  for (int i = 0; i < 4; i++) acc[i] = v8f_zero();

  // cooperative-load assignments
  const int arow = t >> 1;            // A: 128 rows, 2 threads/row
  const int acol = (t & 1) * 16;      // 16 bf16 each
  const int bk = t & 31;              // B: fixed k per thread
  const int bn = (t >> 5) * 8;        // 8 consecutive n each

  auto load_tile = [&](int kt, int buf) {
    // A tile: straight copy, 32B per thread
    const bf16* ga = A + (size_t)(m0 + arow) * K + kt + acol;
    copy16_g2l(&sA[buf][arow * 32 + acol], ga);
    copy16_g2l(&sA[buf][arow * 32 + acol + 8], ga + 8);
    // B tile: transpose [k][n] -> sB[n][k]
    const bf16* gb = B + (size_t)(kt + bk) * N + n0 + bn;
    u32x4 raw = *(const u32x4*)gb;    // 8 bf16
    const bf16* rb = (const bf16*)&raw;
#pragma unroll
    for (int i = 0; i < 8; i++) sB[buf][(bn + i) * 32 + bk] = rb[i];
  };

  load_tile(0, 0);
  const int nsteps = K >> 5;
  for (int i = 0; i < nsteps; ++i) {
    const int cur = i & 1;
    wait_async_lds();
    __syncthreads();
    if (i + 1 < nsteps) load_tile((i + 1) * 32, cur ^ 1);

    // A fragment (16x32 bf16 layout): row = w*16+l16, K halves by lane half
    Frag16 af;
    const bf16* ap = &sA[cur][(w * 16 + l16) * 32];
    af.q[0] = *(const u32x4*)(ap + 16 * half);
    af.q[1] = *(const u32x4*)(ap + 16 * half + 8);

#pragma unroll
    for (int nt = 0; nt < 4; nt++) {
      Frag16 bfr;
      const bf16* bp = &sB[cur][(nt * 16 + l16) * 32 + 16 * half];
      bfr.q[0] = *(const u32x4*)bp;
      bfr.q[1] = *(const u32x4*)(bp + 8);
      acc[nt] = wmma_bf16(af, bfr, acc[nt]);
    }
  }

  // epilogue: C layout M = r + 8*half, N = l16
#pragma unroll
  for (int nt = 0; nt < 4; nt++) {
    const int n = n0 + nt * 16 + l16;
    const float bv = bias[n];
#pragma unroll
    for (int r = 0; r < 8; r++) {
      const int m = m0 + w * 16 + r + 8 * half;
      const float val = acc[nt][r] + bv;
      if (OUT_F32)
        ((float*)outv)[(size_t)m * N + n] = val;
      else
        ((bf16*)outv)[(size_t)m * N + n] = (bf16)val;
    }
  }
}

// ---------------------------------------------------------------------------
// Stage 2: flash attention.
// Grid: (LQ/64, BATCH*H). Block: 128 threads = 4 waves, wave = 16 q-rows.
// Double-buffered K/V tiles; K via async-to-LDS, V transposed to [d][lk].
// Per 32-key step: QK^T via 4 WMMAs, online softmax, PV via 4 WMMAs.
// ---------------------------------------------------------------------------
__global__ __launch_bounds__(128) void flash_attn(
    const bf16* __restrict__ Qm, const bf16* __restrict__ Km,
    const bf16* __restrict__ Vm, bf16* __restrict__ ctx) {
  __shared__ bf16 sK[2][32 * 64];     // [key row][d]   row-major copy
  __shared__ bf16 sV[2][64 * 32];     // [d][key row]   transposed
  __shared__ bf16 sP[4][16 * 32];     // per-wave P scratch (C->A relayout)

  const int t = threadIdx.x;
  const int w = t >> 5;
  const int lane = t & 31;
  const int half = lane >> 4;
  const int l16 = lane & 15;
  const int bh = blockIdx.y;
  const int b = bh >> 4;              // H == 16
  const int h = bh & 15;
  const int qbase = blockIdx.x * 64 + w * 16;

  // ---- load Q fragments for the wave's 16 rows (2 K-steps over D=64),
  //      pre-scaled by 1/sqrt(D)=0.125 (exact power of two in bf16).
  Frag16 qf[2];
  {
    const bf16* qp = Qm + (size_t)(b * LQ + qbase + l16) * AD + h * D;
#pragma unroll
    for (int c = 0; c < 2; c++) {
      qf[c].q[0] = *(const u32x4*)(qp + c * 32 + 16 * half);
      qf[c].q[1] = *(const u32x4*)(qp + c * 32 + 16 * half + 8);
#pragma unroll
      for (int i = 0; i < 16; i++)
        qf[c].v[i] = (bf16)((float)qf[c].v[i] * 0.125f);
    }
  }

  v8f acc[4];
#pragma unroll
  for (int i = 0; i < 4; i++) acc[i] = v8f_zero();
  float mrun[8], lrun[8];
#pragma unroll
  for (int r = 0; r < 8; r++) { mrun[r] = -1e30f; lrun[r] = 0.0f; }

  // cooperative-load assignments
  const int kr4 = t >> 2;             // K copy: 32 rows, 4 thr/row
  const int d16 = (t & 3) * 16;
  const int vr = t & 31;              // V transpose: fixed key row
  const int vd = (t >> 5) * 16;       // 16 d-values each

  auto load_tile = [&](int kt, int buf) {
    // K tile: straight copy (async when available)
    const bf16* gk = Km + (size_t)(b * LK + kt + kr4) * AD + h * D + d16;
    copy16_g2l(&sK[buf][kr4 * 64 + d16], gk);
    copy16_g2l(&sK[buf][kr4 * 64 + d16 + 8], gk + 8);
    // V tile: transpose [lk][d] -> sV[d][lk]
    const bf16* gv = Vm + (size_t)(b * LK + kt + vr) * AD + h * D + vd;
    u32x4 r0 = *(const u32x4*)gv;
    u32x4 r1 = *(const u32x4*)(gv + 8);
    const bf16* rv = (const bf16*)&r0;
#pragma unroll
    for (int i = 0; i < 8; i++) sV[buf][(vd + i) * 32 + vr] = rv[i];
    rv = (const bf16*)&r1;
#pragma unroll
    for (int i = 0; i < 8; i++) sV[buf][(vd + 8 + i) * 32 + vr] = rv[i];
  };

  load_tile(0, 0);
  const int nsteps = LK >> 5;
  for (int i = 0; i < nsteps; ++i) {
    const int cur = i & 1;
    wait_async_lds();
    __syncthreads();
    if (i + 1 < nsteps) load_tile((i + 1) * 32, cur ^ 1);

    // ---- scores S = (Q*scale) @ K^T  (two 16x16 tiles over the 32 keys)
    v8f s0 = v8f_zero(), s1 = v8f_zero();
#pragma unroll
    for (int c = 0; c < 2; c++) {
      Frag16 b0, b1;
      const bf16* p0 = &sK[cur][l16 * 64 + c * 32 + 16 * half];
      const bf16* p1 = &sK[cur][(16 + l16) * 64 + c * 32 + 16 * half];
      b0.q[0] = *(const u32x4*)p0; b0.q[1] = *(const u32x4*)(p0 + 8);
      b1.q[0] = *(const u32x4*)p1; b1.q[1] = *(const u32x4*)(p1 + 8);
      s0 = wmma_bf16(qf[c], b0, s0);
      s1 = wmma_bf16(qf[c], b1, s1);
    }

    // ---- online softmax (row = r + 8*half; n spread across 16-lane group)
#pragma unroll
    for (int r = 0; r < 8; r++) {
      float v0 = s0[r];
      float v1 = s1[r];
      float mx = fmaxf(v0, v1);
      mx = fmaxf(mx, __shfl_xor(mx, 1, 32));
      mx = fmaxf(mx, __shfl_xor(mx, 2, 32));
      mx = fmaxf(mx, __shfl_xor(mx, 4, 32));
      mx = fmaxf(mx, __shfl_xor(mx, 8, 32));
      float mnew = fmaxf(mrun[r], mx);
      float alpha = __expf(mrun[r] - mnew);
      mrun[r] = mnew;
      float p0 = __expf(v0 - mnew);
      float p1 = __expf(v1 - mnew);
      float ps = p0 + p1;
      ps += __shfl_xor(ps, 1, 32);
      ps += __shfl_xor(ps, 2, 32);
      ps += __shfl_xor(ps, 4, 32);
      ps += __shfl_xor(ps, 8, 32);
      lrun[r] = lrun[r] * alpha + ps;
      acc[0][r] *= alpha;
      acc[1][r] *= alpha;
      acc[2][r] *= alpha;
      acc[3][r] *= alpha;
      // write P (C layout -> LDS row-major [m][32])
      sP[w][(r + 8 * half) * 32 + l16]      = (bf16)p0;
      sP[w][(r + 8 * half) * 32 + 16 + l16] = (bf16)p1;
    }
    asm volatile("" ::: "memory");  // keep P writes before A-layout reads

    // ---- re-read P as A fragment (16x32 bf16); wave-private, in-order LDS
    Frag16 pf;
    pf.q[0] = *(const u32x4*)&sP[w][l16 * 32 + 16 * half];
    pf.q[1] = *(const u32x4*)&sP[w][l16 * 32 + 16 * half + 8];

    // ---- ctx += P @ V  (N = d, 4 tiles of 16)
#pragma unroll
    for (int nt = 0; nt < 4; nt++) {
      Frag16 bvf;
      const bf16* vp = &sV[cur][(nt * 16 + l16) * 32 + 16 * half];
      bvf.q[0] = *(const u32x4*)vp;
      bvf.q[1] = *(const u32x4*)(vp + 8);
      acc[nt] = wmma_bf16(pf, bvf, acc[nt]);
    }
  }

  // ---- normalize + store ctx (bf16, [B*LQ][AD])
  float inv[8];
#pragma unroll
  for (int r = 0; r < 8; r++) inv[r] = 1.0f / lrun[r];
#pragma unroll
  for (int nt = 0; nt < 4; nt++) {
#pragma unroll
    for (int r = 0; r < 8; r++) {
      const int m = qbase + r + 8 * half;
      const float val = acc[nt][r] * inv[r];
      ctx[(size_t)(b * LQ + m) * AD + h * D + nt * 16 + l16] = (bf16)val;
    }
  }
}

// ---------------------------------------------------------------------------
// Host-side launch
// ---------------------------------------------------------------------------
extern "C" void kernel_launch(void* const* d_in, const int* in_sizes, int n_in,
                              void* d_out, int out_size, void* d_ws,
                              size_t ws_size, hipStream_t stream) {
  const float* x      = (const float*)d_in[0];
  const float* embeds = (const float*)d_in[1];
  const float* Wq = (const float*)d_in[2]; const float* bq = (const float*)d_in[3];
  const float* Wk = (const float*)d_in[4]; const float* bk = (const float*)d_in[5];
  const float* Wv = (const float*)d_in[6]; const float* bv = (const float*)d_in[7];
  const float* Wo = (const float*)d_in[8]; const float* bo = (const float*)d_in[9];
  float* out = (float*)d_out;

  const size_t NX = (size_t)BATCH * LQ * DIM;   // 8.4M activations
  const size_t NW = (size_t)DIM * AD;           // 1M weights

  char* ws = (char*)d_ws;
  size_t off = 0;
  auto carve = [&](size_t bytes) { void* p = ws + off; off += bytes; return p; };
  bf16* xb   = (bf16*)carve(NX * 2);
  bf16* eb   = (bf16*)carve(NX * 2);
  bf16* wqb  = (bf16*)carve(NW * 2);
  bf16* wkb  = (bf16*)carve(NW * 2);
  bf16* wvb  = (bf16*)carve(NW * 2);
  bf16* wob  = (bf16*)carve(NW * 2);
  bf16* Qb   = (bf16*)carve(NX * 2);
  bf16* Kb   = (bf16*)carve(NX * 2);
  bf16* Vb   = (bf16*)carve(NX * 2);
  bf16* ctxb = (bf16*)carve(NX * 2);

  // Stage 0: conversions
  auto cvt = [&](const float* src, bf16* dst, size_t n) {
    int n4 = (int)(n / 4);
    int blocks = (n4 + 255) / 256;
    if (blocks > 4096) blocks = 4096;
    cvt_f32_to_bf16<<<blocks, 256, 0, stream>>>((const f32x4*)src,
                                                (bf16x4*)dst, n4);
  };
  cvt(x, xb, NX);
  cvt(embeds, eb, NX);
  cvt(Wq, wqb, NW);
  cvt(Wk, wkb, NW);
  cvt(Wv, wvb, NW);
  cvt(Wo, wob, NW);

  // Stage 1: Q/K/V projections (M=8192, N=1024, K=1024), bf16 out
  const int M = BATCH * LQ, N = AD, K = DIM;
  dim3 ggrid(N / 64, M / 128);
  gemm_bf16<false><<<ggrid, 256, 0, stream>>>(xb, wqb, bq, Qb, M, N, K);
  gemm_bf16<false><<<ggrid, 256, 0, stream>>>(eb, wkb, bk, Kb, M, N, K);
  gemm_bf16<false><<<ggrid, 256, 0, stream>>>(eb, wvb, bv, Vb, M, N, K);

  // Stage 2: flash attention
  dim3 fgrid(LQ / 64, BATCH * H);
  flash_attn<<<fgrid, 128, 0, stream>>>(Qb, Kb, Vb, ctxb);

  // Stage 3: output projection (f32 out)
  gemm_bf16<true><<<ggrid, 256, 0, stream>>>(ctxb, wob, bo, (void*)out, M, DIM, AD);
}
